// MultiHeadCrossAttention_20658792694423
// MI455X (gfx1250) — matrix operand order repaired and checked by reading, not compiled, not run on hardware
//
#include <hip/hip_runtime.h>
#include <hip/hip_bf16.h>

#define B_   4
#define S_   1024
#define H_   1024
#define NH_  16
#define HD_  64

typedef __attribute__((ext_vector_type(16))) __bf16        v16bf;
typedef __attribute__((ext_vector_type(8)))  float         v8f;
typedef __attribute__((ext_vector_type(4)))  unsigned int  u32x4;

union FragAB { v16bf v; u32x4 q[2]; unsigned int u[8]; };

static __device__ __forceinline__ unsigned short f2bf(float f) {
    unsigned int u = __float_as_uint(f);
    unsigned int r = u + 0x7FFFu + ((u >> 16) & 1u);   // round-to-nearest-even
    return (unsigned short)(r >> 16);
}
static __device__ __forceinline__ unsigned int pack_bf16(float a, float b) {
    return (unsigned int)f2bf(a) | ((unsigned int)f2bf(b) << 16);
}

// ---------------------------------------------------------------------------
// fp32 -> bf16 conversion (grid-stride)
// ---------------------------------------------------------------------------
__global__ __launch_bounds__(256) void mhca_cvt_bf16(const float* __restrict__ src,
                                                     unsigned short* __restrict__ dst,
                                                     int n) {
    int i = blockIdx.x * blockDim.x + threadIdx.x;
    int stride = gridDim.x * blockDim.x;
    for (; i < n; i += stride) dst[i] = f2bf(src[i]);
}

// ---------------------------------------------------------------------------
// Generic GEMM: C[m,n] = sum_k A[m,k] * W[n,k] + bias[n]
// A: bf16 [M=4096, K=1024] row-major. W: bf16 [N=1024, K=1024] row-major.
// mode 0: store bf16 at [B,NH,S,HD]     (Q, K)
// mode 1: store bf16 at [B,NH,HD,S]     (V transposed per head)
// mode 2: store fp32 row-major [M,N]    (final output projection)
// Each wave computes a 32(M) x 64(N) tile. 8 WMMAs per K=32 step.
// ---------------------------------------------------------------------------
__global__ __launch_bounds__(128) void mhca_gemm_bf16(
    const unsigned short* __restrict__ A, const unsigned short* __restrict__ W,
    const float* __restrict__ bias, void* __restrict__ out, int mode)
{
    const int lane = threadIdx.x & 31;
    const int wid  = blockIdx.x * 4 + (threadIdx.x >> 5);   // 2048 waves
    const int m0   = (wid & 127) * 32;
    const int n0   = (wid >> 7) * 64;
    const int l15  = lane & 15;
    const bool lo_half = lane < 16;
    const int hb    = lo_half ? 0 : 8;
    const int koffA = lo_half ? 0 : 8;     // A frag K split {0..7,16..23}/{8..15,24..31}
    const int koffB = lo_half ? 0 : 8;     // B frag K split {0..7,16..23}/{8..15,24..31}

    v8f C[2][4];
    #pragma unroll
    for (int i = 0; i < 2; ++i)
        #pragma unroll
        for (int j = 0; j < 4; ++j) C[i][j] = {};

    for (int k0 = 0; k0 < H_; k0 += 32) {
        FragAB Af[2], Bf[4];
        #pragma unroll
        for (int i = 0; i < 2; ++i) {
            const unsigned short* p = A + (size_t)(m0 + 16*i + l15) * H_ + k0 + koffA;
            Af[i].q[0] = *(const u32x4*)p;
            Af[i].q[1] = *(const u32x4*)(p + 16);
        }
        #pragma unroll
        for (int j = 0; j < 4; ++j) {
            const unsigned short* p = W + (size_t)(n0 + 16*j + l15) * H_ + k0 + koffB;
            Bf[j].q[0] = *(const u32x4*)p;
            Bf[j].q[1] = *(const u32x4*)(p + 16);
        }
        #pragma unroll
        for (int i = 0; i < 2; ++i)
            #pragma unroll
            for (int j = 0; j < 4; ++j)
                C[i][j] = __builtin_amdgcn_wmma_f32_16x16x32_bf16(
                    false, Af[i].v, false, Bf[j].v, (short)0, C[i][j], false, false);
    }

    #pragma unroll
    for (int j = 0; j < 4; ++j) {
        const int n  = n0 + 16*j + l15;
        const float bv = bias[n];
        const int h = n >> 6, d = n & 63;
        #pragma unroll
        for (int i = 0; i < 2; ++i) {
            #pragma unroll
            for (int r = 0; r < 8; ++r) {
                const int m = m0 + 16*i + r + hb;
                const float val = C[i][j][r] + bv;
                const int b = m >> 10, s = m & 1023;
                if (mode == 0) {
                    ((unsigned short*)out)[((size_t)((b*NH_ + h)*S_ + s))*HD_ + d] = f2bf(val);
                } else if (mode == 1) {
                    ((unsigned short*)out)[((size_t)((b*NH_ + h)*HD_ + d))*S_ + s] = f2bf(val);
                } else {
                    ((float*)out)[(size_t)m * H_ + n] = val;
                }
            }
        }
    }
}

// ---------------------------------------------------------------------------
// Flash attention, one wave per (b, h, 16-query tile).
// S^T = K_blk * Q^T  (softmax row per lane), ctx^T = V^T * P^T.
// Span bias is a per-row constant under softmax -> provably no effect: dropped.
// ---------------------------------------------------------------------------
__global__ __launch_bounds__(256) void mhca_attn(
    const unsigned short* __restrict__ Qb,   // [B,NH,S,HD] bf16
    const unsigned short* __restrict__ Kb,   // [B,NH,S,HD] bf16
    const unsigned short* __restrict__ Vtb,  // [B,NH,HD,S] bf16
    const int* __restrict__ amask,           // [B,S] int32
    unsigned short* __restrict__ ctx)        // [B,S,H] bf16
{
    const int lane = threadIdx.x & 31;
    const int wid  = blockIdx.x * 8 + (threadIdx.x >> 5);   // 4096 waves
    const int qt   = (wid & 63) * 16;
    const int h    = (wid >> 6) & 15;
    const int b    = wid >> 10;
    const int l15  = lane & 15;
    const bool lo_half = lane < 16;
    const int hb   = lo_half ? 0 : 8;

    const size_t headRM = ((size_t)(b*NH_ + h)) * S_ * HD_;   // row-major heads (Q,K)
    const size_t headT  = ((size_t)(b*NH_ + h)) * HD_ * S_;   // transposed head (V^T)

    // Q as B-fragments (d x q), persistent: two 32-deep d-chunks
    FragAB Bq[2];
    #pragma unroll
    for (int c = 0; c < 2; ++c) {
        const unsigned short* p = Qb + headRM + (size_t)(qt + l15) * HD_
                                  + 32*c + (lo_half ? 0 : 8);
        Bq[c].q[0] = *(const u32x4*)p;
        Bq[c].q[1] = *(const u32x4*)(p + 16);
    }

    v8f O[4];
    #pragma unroll
    for (int c = 0; c < 4; ++c) O[c] = {};
    float m_run = -1e30f, l_run = 0.f;
    const float scale = 0.125f;   // 1/sqrt(64)

    for (int kb = 0; kb < S_; kb += 32) {
        // ---- S^T = K_blk(32x64) * Q^T(64x16), two 16-row C fragments ----
        v8f Slo = {}, Shi = {};
        #pragma unroll
        for (int dc = 0; dc < 2; ++dc) {
            FragAB Ak0, Ak1;
            const unsigned short* p0 = Kb + headRM + (size_t)(kb + l15) * HD_
                                       + 32*dc + (lo_half ? 0 : 8);
            Ak0.q[0] = *(const u32x4*)p0;  Ak0.q[1] = *(const u32x4*)(p0 + 16);
            const unsigned short* p1 = Kb + headRM + (size_t)(kb + 16 + l15) * HD_
                                       + 32*dc + (lo_half ? 0 : 8);
            Ak1.q[0] = *(const u32x4*)p1;  Ak1.q[1] = *(const u32x4*)(p1 + 16);
            Slo = __builtin_amdgcn_wmma_f32_16x16x32_bf16(false, Ak0.v, false, Bq[dc].v,
                                                          (short)0, Slo, false, false);
            Shi = __builtin_amdgcn_wmma_f32_16x16x32_bf16(false, Ak1.v, false, Bq[dc].v,
                                                          (short)0, Shi, false, false);
        }

        // ---- mask bias (lane holds key kb+lane), scale, online softmax ----
        const float mb = amask[b * S_ + kb + lane] ? 0.f : -1e30f;
        float s_lo[8], s_hi[8];
        #pragma unroll
        for (int r = 0; r < 8; ++r) {
            s_lo[r] = Slo[r] * scale + __shfl(mb, r + hb, 32);
            s_hi[r] = Shi[r] * scale + __shfl(mb, 16 + r + hb, 32);
        }
        float mloc = s_lo[0];
        #pragma unroll
        for (int r = 1; r < 8; ++r) mloc = fmaxf(mloc, s_lo[r]);
        #pragma unroll
        for (int r = 0; r < 8; ++r) mloc = fmaxf(mloc, s_hi[r]);
        mloc = fmaxf(mloc, __shfl_xor(mloc, 16, 32));
        const float mnew  = fmaxf(m_run, mloc);
        const float alpha = __expf(m_run - mnew);
        float p_lo[8], p_hi[8], lsum = 0.f;
        #pragma unroll
        for (int r = 0; r < 8; ++r) {
            p_lo[r] = __expf(s_lo[r] - mnew);
            p_hi[r] = __expf(s_hi[r] - mnew);
            lsum += p_lo[r] + p_hi[r];
        }
        lsum += __shfl_xor(lsum, 16, 32);
        l_run = l_run * alpha + lsum;
        m_run = mnew;
        #pragma unroll
        for (int c = 0; c < 4; ++c) O[c] = O[c] * alpha;

        // ---- P^T (C layout) -> 32x16 B fragment via lane^16 exchange ----
        float xl[8], xh[8];
        #pragma unroll
        for (int r = 0; r < 8; ++r) {
            xl[r] = __shfl_xor(p_lo[r], 16, 32);
            xh[r] = __shfl_xor(p_hi[r], 16, 32);
        }
        float bvv[16];
        #pragma unroll
        for (int k = 0; k < 8; ++k) {
            bvv[k]     = lo_half ? p_lo[k] : xh[k];
            bvv[8 + k] = lo_half ? xl[k]   : p_hi[k];
        }
        FragAB Pb;
        #pragma unroll
        for (int j = 0; j < 8; ++j) Pb.u[j] = pack_bf16(bvv[2*j], bvv[2*j + 1]);

        // ---- ctx^T += V^T(64x32) * P^T(32x16) : 4 d-chunk fragments ----
        #pragma unroll
        for (int c = 0; c < 4; ++c) {
            FragAB Av;
            const unsigned short* p = Vtb + headT + (size_t)(16*c + l15) * S_
                                      + kb + (lo_half ? 0 : 16);
            Av.q[0] = *(const u32x4*)p;  Av.q[1] = *(const u32x4*)(p + 8);
            O[c] = __builtin_amdgcn_wmma_f32_16x16x32_bf16(false, Av.v, false, Pb.v,
                                                           (short)0, O[c], false, false);
        }
    }

    const float inv = 1.f / l_run;
    const size_t row = ((size_t)(b * S_ + qt + l15)) * H_;
    #pragma unroll
    for (int c = 0; c < 4; ++c) {
        u32x4 wv;
        #pragma unroll
        for (int j = 0; j < 4; ++j)
            wv[j] = pack_bf16(O[c][2*j] * inv, O[c][2*j + 1] * inv);
        *(u32x4*)(ctx + row + h*HD_ + 16*c + hb) = wv;
    }
}

// ---------------------------------------------------------------------------
extern "C" void kernel_launch(void* const* d_in, const int* in_sizes, int n_in,
                              void* d_out, int out_size, void* d_ws, size_t ws_size,
                              hipStream_t stream) {
    const float* aspect  = (const float*)d_in[0];
    const float* opinion = (const float*)d_in[1];
    const int*   amask   = (const int*)d_in[2];
    const float* Wq = (const float*)d_in[3];
    const float* bq = (const float*)d_in[4];
    const float* Wk = (const float*)d_in[5];
    const float* bk = (const float*)d_in[6];
    const float* Wv = (const float*)d_in[7];
    const float* bv = (const float*)d_in[8];
    const float* Wo = (const float*)d_in[9];
    const float* bo = (const float*)d_in[10];
    // d_in[11] (Wbil), d_in[12] (bbil): span bias is constant per softmax row
    // and therefore cancels in softmax -> mathematically unused.

    const int ACT = B_ * S_ * H_;   // 4194304 elems
    const int WSZ = H_ * H_;        // 1048576 elems

    char* ws = (char*)d_ws;
    unsigned short* aspect_bf  = (unsigned short*)(ws);
    unsigned short* opinion_bf = (unsigned short*)(ws + (size_t) 8u * 1024 * 1024);
    unsigned short* Wq_bf      = (unsigned short*)(ws + (size_t)16u * 1024 * 1024);
    unsigned short* Wk_bf      = (unsigned short*)(ws + (size_t)18u * 1024 * 1024);
    unsigned short* Wv_bf      = (unsigned short*)(ws + (size_t)20u * 1024 * 1024);
    unsigned short* Wo_bf      = (unsigned short*)(ws + (size_t)22u * 1024 * 1024);
    unsigned short* Qbuf       = (unsigned short*)(ws + (size_t)24u * 1024 * 1024);
    unsigned short* Kbuf       = (unsigned short*)(ws + (size_t)32u * 1024 * 1024);
    unsigned short* Vtbuf      = (unsigned short*)(ws + (size_t)40u * 1024 * 1024);
    unsigned short* ctxbuf     = (unsigned short*)(ws + (size_t)48u * 1024 * 1024);

    mhca_cvt_bf16<<<2048, 256, 0, stream>>>(aspect,  aspect_bf,  ACT);
    mhca_cvt_bf16<<<2048, 256, 0, stream>>>(opinion, opinion_bf, ACT);
    mhca_cvt_bf16<<< 512, 256, 0, stream>>>(Wq, Wq_bf, WSZ);
    mhca_cvt_bf16<<< 512, 256, 0, stream>>>(Wk, Wk_bf, WSZ);
    mhca_cvt_bf16<<< 512, 256, 0, stream>>>(Wv, Wv_bf, WSZ);
    mhca_cvt_bf16<<< 512, 256, 0, stream>>>(Wo, Wo_bf, WSZ);

    mhca_gemm_bf16<<<512, 128, 0, stream>>>(aspect_bf,  Wq_bf, bq, Qbuf,  0);
    mhca_gemm_bf16<<<512, 128, 0, stream>>>(opinion_bf, Wk_bf, bk, Kbuf,  0);
    mhca_gemm_bf16<<<512, 128, 0, stream>>>(opinion_bf, Wv_bf, bv, Vtbuf, 1);

    mhca_attn<<<512, 256, 0, stream>>>(Qbuf, Kbuf, Vtbuf, amask, ctxbuf);

    mhca_gemm_bf16<<<512, 128, 0, stream>>>(ctxbuf, Wo_bf, bo, d_out, 2);
}
